// DCRNCognition_75196287418381
// MI455X (gfx1250) — matrix operand order repaired and checked by reading, not compiled, not run on hardware
//
#include <hip/hip_runtime.h>
#include <hip/hip_bf16.h>
#include <math.h>

typedef __bf16 bf16;
typedef __attribute__((ext_vector_type(16))) __bf16 v16bf;
typedef __attribute__((ext_vector_type(8)))  __bf16 v8bf;
typedef __attribute__((ext_vector_type(8)))  float  v8f;

#define T_MAX 512
#define BATCH 128
#define D_IN  256
#define NCLS  7
#define STEPS 2

// ---------------- WMMA fragment helpers (ISA 7.12.2 layouts) ----------------
// A (16x32 bf16): lane m = lane%16, h = lane/16; elems 0..7 = K[k0+8h .. +7],
//                 elems 8..15 = K[k0+16+8h .. +7]
__device__ __forceinline__ v16bf ld_frag_a(const bf16* __restrict__ base,
                                           size_t stride, int k0) {
  int lane = threadIdx.x & 31;
  const bf16* p = base + (size_t)(lane & 15) * stride + (size_t)k0 + ((lane >> 4) << 3);
  v8bf lo = *(const v8bf*)p;
  v8bf hi = *(const v8bf*)(p + 16);
  return __builtin_shufflevector(lo, hi, 0,1,2,3,4,5,6,7,8,9,10,11,12,13,14,15);
}
// B (32x16 bf16): lane n = lane%16, h = lane/16; elems 0..7 = K[k0+8h .. +7],
//                 elems 8..15 = K[k0+16+8h .. +7]
__device__ __forceinline__ v16bf ld_frag_b(const bf16* __restrict__ base,
                                           size_t stride, int k0) {
  int lane = threadIdx.x & 31;
  const bf16* p = base + (size_t)(lane & 15) * stride + (size_t)k0 + ((lane >> 4) << 3);
  v8bf lo = *(const v8bf*)p;
  v8bf hi = *(const v8bf*)(p + 16);
  return __builtin_shufflevector(lo, hi, 0,1,2,3,4,5,6,7,8,9,10,11,12,13,14,15);
}
__device__ __forceinline__ v8f wmma_bf16(v16bf a, v16bf b, v8f c) {
  return __builtin_amdgcn_wmma_f32_16x16x32_bf16(false, a, false, b, (short)0, c,
                                                 false, false);
}
__device__ __forceinline__ float sigmoidf_(float x) { return 1.0f / (1.0f + __expf(-x)); }

#define V8F_ZERO {0.f,0.f,0.f,0.f,0.f,0.f,0.f,0.f}

// ---------------- prep: mask + f32->bf16 bank + transposed bank -------------
__global__ void k_prep(const float* __restrict__ src, const int* __restrict__ len,
                       bf16* __restrict__ bank, bf16* __restrict__ bankT) {
  size_t idx = (size_t)blockIdx.x * blockDim.x + threadIdx.x;
  const size_t total = (size_t)T_MAX * BATCH * D_IN;
  if (idx >= total) return;
  int d = (int)(idx % D_IN);
  size_t tb = idx / D_IN;
  int b = (int)(tb % BATCH);
  int t = (int)(tb / BATCH);
  float v = (t < len[b]) ? src[idx] : 0.0f;
  bf16 bv = (bf16)v;
  bank[idx] = bv;                                    // [T*B, D]
  bankT[((size_t)b * D_IN + d) * T_MAX + t] = bv;    // [B, D, U]
}

__global__ void k_cvt(const float* __restrict__ s, bf16* __restrict__ d, int n) {
  int i = blockIdx.x * blockDim.x + threadIdx.x;
  if (i < n) d[i] = (bf16)s[i];
}

// -------- fc: qstar[:,0:2D] = bank @ fc_w^T + fc_b (32x32 tile / wave) ------
__global__ void k_fc(const bf16* __restrict__ A, const bf16* __restrict__ W,
                     const float* __restrict__ bias, bf16* __restrict__ Q) {
  const int N = 2 * D_IN, K = D_IN;
  int wave = blockIdx.x * (blockDim.x >> 5) + (threadIdx.x >> 5);
  int nt2 = wave & (N / 32 - 1);          // 16 col-pairs
  int mt2 = wave / (N / 32);
  if (mt2 >= (T_MAX * BATCH) / 32) return;
  const bf16* A0 = A + (size_t)mt2 * 32 * K;
  const bf16* A1 = A0 + (size_t)16 * K;
  const bf16* B0 = W + (size_t)nt2 * 32 * K;
  const bf16* B1 = B0 + (size_t)16 * K;
  v8f c00 = V8F_ZERO, c01 = V8F_ZERO, c10 = V8F_ZERO, c11 = V8F_ZERO;
  for (int k0 = 0; k0 < K; k0 += 32) {
    __builtin_prefetch((const void*)(A0 + (size_t)(threadIdx.x & 15) * K + k0 + 64), 0, 3);
    v16bf a0 = ld_frag_a(A0, K, k0), a1 = ld_frag_a(A1, K, k0);
    v16bf b0 = ld_frag_b(B0, K, k0), b1 = ld_frag_b(B1, K, k0);
    c00 = wmma_bf16(a0, b0, c00); c01 = wmma_bf16(a0, b1, c01);
    c10 = wmma_bf16(a1, b0, c10); c11 = wmma_bf16(a1, b1, c11);
  }
  int lane = threadIdx.x & 31;
  int n0 = nt2 * 32 + (lane & 15), n1 = n0 + 16;
  int mh = (lane >> 4) * 8;
  float bv0 = bias[n0], bv1 = bias[n1];
#pragma unroll
  for (int i = 0; i < 8; ++i) {
    size_t m0 = (size_t)mt2 * 32 + mh + i, m1 = m0 + 16;
    Q[m0 * (size_t)N + n0] = (bf16)(c00[i] + bv0);
    Q[m0 * (size_t)N + n1] = (bf16)(c01[i] + bv1);
    Q[m1 * (size_t)N + n0] = (bf16)(c10[i] + bv0);
    Q[m1 * (size_t)N + n1] = (bf16)(c11[i] + bv1);
  }
}

// -- fused gate GEMMs + LSTM cell: 32 rows x 16 cols x 4 gates per wave ------
__global__ void k_gates(const bf16* __restrict__ qs, const bf16* __restrict__ hin,
                        bf16* __restrict__ hout, float* __restrict__ c,
                        const bf16* __restrict__ wih, const bf16* __restrict__ whh,
                        const float* __restrict__ bih, const float* __restrict__ bhh,
                        int first) {
  const int D = D_IN, K1 = 2 * D_IN;
  int wave = blockIdx.x * (blockDim.x >> 5) + (threadIdx.x >> 5);
  int mt2 = wave >> 4, dt = wave & 15;
  if (mt2 >= (T_MAX * BATCH) / 32) return;
  v8f g0[4] = {V8F_ZERO, V8F_ZERO, V8F_ZERO, V8F_ZERO};  // rows m..m+15, gates i,f,g,o
  v8f g1[4] = {V8F_ZERO, V8F_ZERO, V8F_ZERO, V8F_ZERO};  // rows m+16..m+31
  {
    const bf16* A0 = qs + (size_t)mt2 * 32 * K1;
    const bf16* A1 = A0 + (size_t)16 * K1;
    const bf16* Bg[4];
#pragma unroll
    for (int g = 0; g < 4; ++g) Bg[g] = wih + (size_t)(g * D + dt * 16) * K1;
    for (int k0 = 0; k0 < K1; k0 += 32) {
      __builtin_prefetch((const void*)(A0 + (size_t)(threadIdx.x & 15) * K1 + k0 + 64), 0, 3);
      v16bf a0 = ld_frag_a(A0, K1, k0), a1 = ld_frag_a(A1, K1, k0);
#pragma unroll
      for (int g = 0; g < 4; ++g) {
        v16bf b = ld_frag_b(Bg[g], K1, k0);
        g0[g] = wmma_bf16(a0, b, g0[g]);
        g1[g] = wmma_bf16(a1, b, g1[g]);
      }
    }
  }
  if (!first) {
    const bf16* A0 = hin + (size_t)mt2 * 32 * D;
    const bf16* A1 = A0 + (size_t)16 * D;
    const bf16* Bg[4];
#pragma unroll
    for (int g = 0; g < 4; ++g) Bg[g] = whh + (size_t)(g * D + dt * 16) * D;
    for (int k0 = 0; k0 < D; k0 += 32) {
      v16bf a0 = ld_frag_a(A0, D, k0), a1 = ld_frag_a(A1, D, k0);
#pragma unroll
      for (int g = 0; g < 4; ++g) {
        v16bf b = ld_frag_b(Bg[g], D, k0);
        g0[g] = wmma_bf16(a0, b, g0[g]);
        g1[g] = wmma_bf16(a1, b, g1[g]);
      }
    }
  }
  int lane = threadIdx.x & 31;
  int n = lane & 15, mh = (lane >> 4) * 8;
  int d = dt * 16 + n;
  float bi = bih[d] + bhh[d];
  float bff = bih[D + d] + bhh[D + d];
  float bg = bih[2 * D + d] + bhh[2 * D + d];
  float bo = bih[3 * D + d] + bhh[3 * D + d];
#pragma unroll
  for (int half = 0; half < 2; ++half) {
    v8f* G = half ? g1 : g0;
#pragma unroll
    for (int i = 0; i < 8; ++i) {
      size_t row = (size_t)mt2 * 32 + half * 16 + mh + i;
      float iv = sigmoidf_(G[0][i] + bi);
      float fv = sigmoidf_(G[1][i] + bff);
      float gv = tanhf(G[2][i] + bg);
      float ov = sigmoidf_(G[3][i] + bo);
      float co = first ? 0.0f : c[row * D + d];
      float cn = fv * co + iv * gv;
      float hv = ov * tanhf(cn);
      c[row * D + d] = cn;
      hout[row * D + d] = (bf16)hv;
    }
  }
}

__global__ void k_copyh(const bf16* __restrict__ h, bf16* __restrict__ qs) {
  size_t idx = (size_t)blockIdx.x * blockDim.x + threadIdx.x;
  const size_t total = (size_t)T_MAX * BATCH * D_IN;
  if (idx >= total) return;
  size_t row = idx / D_IN;
  int d = (int)(idx % D_IN);
  qs[row * (2 * D_IN) + d] = h[idx];
}

// --- attention scores e[b,t,u] = h[t,b,:] . bank[u,b,:] (32x32 tile/wave) ---
__global__ void k_e(const bf16* __restrict__ h, const bf16* __restrict__ bank,
                    const int* __restrict__ len, float* __restrict__ e) {
  const int D = D_IN, T = T_MAX, B = BATCH;
  int wave = blockIdx.x * (blockDim.x >> 5) + (threadIdx.x >> 5);
  int b = wave >> 8;                       // 16*16 32x32-tiles per batch
  if (b >= B) return;
  int rem = wave & 255;
  int tt2 = rem >> 4, ut2 = rem & 15;
  size_t strd = (size_t)B * D;
  const bf16* A0 = h + ((size_t)tt2 * 32 * B + b) * D;
  const bf16* A1 = A0 + 16 * strd;
  const bf16* B0 = bank + ((size_t)ut2 * 32 * B + b) * D;
  const bf16* B1 = B0 + 16 * strd;
  v8f c00 = V8F_ZERO, c01 = V8F_ZERO, c10 = V8F_ZERO, c11 = V8F_ZERO;
  for (int k0 = 0; k0 < D; k0 += 32) {
    __builtin_prefetch((const void*)(A0 + (size_t)(threadIdx.x & 15) * strd + k0 + 64), 0, 3);
    v16bf a0 = ld_frag_a(A0, strd, k0), a1 = ld_frag_a(A1, strd, k0);
    v16bf b0 = ld_frag_b(B0, strd, k0), b1 = ld_frag_b(B1, strd, k0);
    c00 = wmma_bf16(a0, b0, c00); c01 = wmma_bf16(a0, b1, c01);
    c10 = wmma_bf16(a1, b0, c10); c11 = wmma_bf16(a1, b1, c11);
  }
  int L = len[b];
  int lane = threadIdx.x & 31;
  int u0 = ut2 * 32 + (lane & 15), u1 = u0 + 16;
  int mh = (lane >> 4) * 8;
  float m0 = (u0 < L) ? 0.0f : 1.0f;  // selector
  float m1 = (u1 < L) ? 0.0f : 1.0f;
#pragma unroll
  for (int i = 0; i < 8; ++i) {
    int t0 = tt2 * 32 + mh + i, t1 = t0 + 16;
    float* r0 = e + ((size_t)b * T + t0) * T;
    float* r1 = e + ((size_t)b * T + t1) * T;
    r0[u0] = m0 ? -1.0e30f : c00[i];
    r0[u1] = m1 ? -1.0e30f : c01[i];
    r1[u0] = m0 ? -1.0e30f : c10[i];
    r1[u1] = m1 ? -1.0e30f : c11[i];
  }
}

// -------- row softmax over U=512 (one wave per row, shfl reductions) --------
__global__ void k_softmax(const float* __restrict__ e, bf16* __restrict__ a) {
  const int U = T_MAX;
  int warp = threadIdx.x >> 5, lane = threadIdx.x & 31;
  size_t row = (size_t)blockIdx.x * (blockDim.x >> 5) + warp;
  if (row >= (size_t)BATCH * T_MAX) return;
  const float* p = e + row * (size_t)U;
  float v[U / 32];
  float mx = -3.0e38f;
#pragma unroll
  for (int j = 0; j < U / 32; ++j) { v[j] = p[lane + 32 * j]; mx = fmaxf(mx, v[j]); }
  for (int off = 16; off; off >>= 1) mx = fmaxf(mx, __shfl_xor(mx, off, 32));
  float s = 0.f;
#pragma unroll
  for (int j = 0; j < U / 32; ++j) { v[j] = __expf(v[j] - mx); s += v[j]; }
  for (int off = 16; off; off >>= 1) s += __shfl_xor(s, off, 32);
  float inv = 1.0f / s;
  bf16* q = a + row * (size_t)U;
#pragma unroll
  for (int j = 0; j < U / 32; ++j) q[lane + 32 * j] = (bf16)(v[j] * inv);
}

// -- r[t,b,:] = a[b,t,:] @ bank[b]; 32x32 tile/wave; into qstar[:, D:2D] -----
__global__ void k_r(const bf16* __restrict__ a, const bf16* __restrict__ bankT,
                    bf16* __restrict__ qs) {
  const int T = T_MAX, B = BATCH, D = D_IN;
  int wave = blockIdx.x * (blockDim.x >> 5) + (threadIdx.x >> 5);
  int b = wave >> 7;                       // 16*8 32x32-tiles per batch
  if (b >= B) return;
  int rem = wave & 127;
  int tt2 = rem >> 3, dt2 = rem & 7;
  const bf16* A0 = a + (size_t)b * T * T + (size_t)tt2 * 32 * T;
  const bf16* A1 = A0 + (size_t)16 * T;
  const bf16* B0 = bankT + (size_t)b * D * T + (size_t)dt2 * 32 * T;
  const bf16* B1 = B0 + (size_t)16 * T;
  v8f c00 = V8F_ZERO, c01 = V8F_ZERO, c10 = V8F_ZERO, c11 = V8F_ZERO;
  for (int k0 = 0; k0 < T; k0 += 32) {
    __builtin_prefetch((const void*)(A0 + (size_t)(threadIdx.x & 15) * T + k0 + 64), 0, 3);
    v16bf a0 = ld_frag_a(A0, T, k0), a1 = ld_frag_a(A1, T, k0);
    v16bf b0 = ld_frag_b(B0, T, k0), b1 = ld_frag_b(B1, T, k0);
    c00 = wmma_bf16(a0, b0, c00); c01 = wmma_bf16(a0, b1, c01);
    c10 = wmma_bf16(a1, b0, c10); c11 = wmma_bf16(a1, b1, c11);
  }
  int lane = threadIdx.x & 31;
  int d0 = dt2 * 32 + (lane & 15), d1 = d0 + 16;
  int mh = (lane >> 4) * 8;
#pragma unroll
  for (int i = 0; i < 8; ++i) {
    int t0 = tt2 * 32 + mh + i, t1 = t0 + 16;
    bf16* r0 = qs + ((size_t)t0 * B + b) * (2 * D) + D;
    bf16* r1 = qs + ((size_t)t1 * B + b) * (2 * D) + D;
    r0[d0] = (bf16)c00[i];
    r0[d1] = (bf16)c01[i];
    r1[d0] = (bf16)c10[i];
    r1[d1] = (bf16)c11[i];
  }
}

// ---------------- output offsets (prefix sum over lengths) ------------------
__global__ void k_offsets(const int* __restrict__ len, int* __restrict__ offs) {
  if (blockIdx.x == 0 && threadIdx.x == 0) {
    int s = 0;
    for (int b = 0; b < BATCH; ++b) { offs[b] = s; s += len[b]; }
    offs[BATCH] = s;
  }
}

// ------- head: logits = relu([qs,qp]) @ out_w^T + b; log_softmax; gather ----
__global__ void k_out(const bf16* __restrict__ qs, const bf16* __restrict__ qp,
                      const float* __restrict__ ow, const float* __restrict__ ob,
                      const int* __restrict__ len, const int* __restrict__ offs,
                      float* __restrict__ out) {
  const int B = BATCH, H = 4 * D_IN, HD = 2 * D_IN;
  int wave = blockIdx.x * (blockDim.x >> 5) + (threadIdx.x >> 5);
  int lane = threadIdx.x & 31;
  if (wave >= T_MAX * BATCH) return;
  int row = wave;
  int t = row / B, b = row % B;
  if (t >= len[b]) return;
  float acc[NCLS] = {0.f, 0.f, 0.f, 0.f, 0.f, 0.f, 0.f};
  const bf16* qrow = qs + (size_t)row * HD;
  const bf16* prow = qp + (size_t)row * HD;
  for (int j = 0; j < H / 32; ++j) {
    int k = lane + 32 * j;
    float hv = (k < HD) ? (float)qrow[k] : (float)prow[k - HD];
    hv = fmaxf(hv, 0.0f);
#pragma unroll
    for (int cc = 0; cc < NCLS; ++cc) acc[cc] += hv * ow[cc * H + k];
  }
#pragma unroll
  for (int cc = 0; cc < NCLS; ++cc)
    for (int off = 16; off; off >>= 1) acc[cc] += __shfl_xor(acc[cc], off, 32);
  if (lane == 0) {
    float l[NCLS], m = -3.0e38f;
#pragma unroll
    for (int cc = 0; cc < NCLS; ++cc) { l[cc] = acc[cc] + ob[cc]; m = fmaxf(m, l[cc]); }
    float s = 0.f;
#pragma unroll
    for (int cc = 0; cc < NCLS; ++cc) s += __expf(l[cc] - m);
    float lse = m + __logf(s);
    float* o = out + ((size_t)offs[b] + t) * NCLS;
#pragma unroll
    for (int cc = 0; cc < NCLS; ++cc) o[cc] = l[cc] - lse;
  }
}

// ---------------------------------------------------------------------------
extern "C" void kernel_launch(void* const* d_in, const int* in_sizes, int n_in,
                              void* d_out, int out_size, void* d_ws, size_t ws_size,
                              hipStream_t stream) {
  (void)in_sizes; (void)n_in; (void)out_size; (void)ws_size;
  const float* input    = (const float*)d_in[0];
  const float* speakers = (const float*)d_in[1];
  const int*   lengths  = (const int*)d_in[2];
  const float* fc_w     = (const float*)d_in[3];
  const float* fc_b     = (const float*)d_in[4];
  const float* w_ih_s   = (const float*)d_in[5];
  const float* w_hh_s   = (const float*)d_in[6];
  const float* b_ih_s   = (const float*)d_in[7];
  const float* b_hh_s   = (const float*)d_in[8];
  const float* w_ih_p   = (const float*)d_in[9];
  const float* w_hh_p   = (const float*)d_in[10];
  const float* b_ih_p   = (const float*)d_in[11];
  const float* b_hh_p   = (const float*)d_in[12];
  const float* out_w    = (const float*)d_in[13];
  const float* out_b    = (const float*)d_in[14];
  float* out = (float*)d_out;

  char* base = (char*)d_ws;
  size_t woff = 0;
  auto alloc = [&](size_t bytes) -> void* {
    void* p = base + woff;
    woff = (woff + bytes + 255) & ~(size_t)255;
    return p;
  };
  bf16* fcw_bf  = (bf16*)alloc((size_t)512 * 256 * 2);
  bf16* wihs_bf = (bf16*)alloc((size_t)1024 * 512 * 2);
  bf16* whhs_bf = (bf16*)alloc((size_t)1024 * 256 * 2);
  bf16* wihp_bf = (bf16*)alloc((size_t)1024 * 512 * 2);
  bf16* whhp_bf = (bf16*)alloc((size_t)1024 * 256 * 2);
  bf16* bank    = (bf16*)alloc((size_t)T_MAX * BATCH * D_IN * 2);
  bf16* bankT   = (bf16*)alloc((size_t)BATCH * D_IN * T_MAX * 2);
  bf16* qs_s    = (bf16*)alloc((size_t)T_MAX * BATCH * 2 * D_IN * 2);
  bf16* qs_p    = (bf16*)alloc((size_t)T_MAX * BATCH * 2 * D_IN * 2);
  bf16* hA      = (bf16*)alloc((size_t)T_MAX * BATCH * D_IN * 2);
  bf16* hB      = (bf16*)alloc((size_t)T_MAX * BATCH * D_IN * 2);
  float* cbuf   = (float*)alloc((size_t)T_MAX * BATCH * D_IN * 4);
  float* ebuf   = (float*)alloc((size_t)BATCH * T_MAX * T_MAX * 4);
  bf16* abuf    = (bf16*)alloc((size_t)BATCH * T_MAX * T_MAX * 2);
  int*  offs    = (int*)alloc((size_t)(BATCH + 1) * 4);

  // weight conversions (tiny)
  k_cvt<<<(512 * 256 + 255) / 256, 256, 0, stream>>>(fc_w, fcw_bf, 512 * 256);
  k_cvt<<<(1024 * 512 + 255) / 256, 256, 0, stream>>>(w_ih_s, wihs_bf, 1024 * 512);
  k_cvt<<<(1024 * 256 + 255) / 256, 256, 0, stream>>>(w_hh_s, whhs_bf, 1024 * 256);
  k_cvt<<<(1024 * 512 + 255) / 256, 256, 0, stream>>>(w_ih_p, wihp_bf, 1024 * 512);
  k_cvt<<<(1024 * 256 + 255) / 256, 256, 0, stream>>>(w_hh_p, whhp_bf, 1024 * 256);

  const int elemBlocks = (int)(((size_t)T_MAX * BATCH * D_IN + 255) / 256);
  const dim3 blk128(128), blk256(256);
  const int fcBlocks   = (T_MAX * BATCH / 32) * (2 * D_IN / 32) / 4;  // 8192
  const int gateBlocks = (T_MAX * BATCH / 32) * (D_IN / 16) / 4;      // 8192
  const int eBlocks    = BATCH * (T_MAX / 32) * (T_MAX / 32) / 4;     // 8192
  const int smBlocks   = (BATCH * T_MAX) / 8;                          // 8192
  const int rBlocks    = BATCH * (T_MAX / 32) * (D_IN / 32) / 4;      // 4096
  const int outBlocks  = (T_MAX * BATCH) / 8;                          // 8192

  struct Stream { const float* src; bf16* wih; bf16* whh; const float* bih; const float* bhh; bf16* qs; };
  Stream streams[2] = {
    { input,    wihs_bf, whhs_bf, b_ih_s, b_hh_s, qs_s },
    { speakers, wihp_bf, whhp_bf, b_ih_p, b_hh_p, qs_p },
  };

  for (int si = 0; si < 2; ++si) {
    Stream& S = streams[si];
    k_prep<<<elemBlocks, blk256, 0, stream>>>(S.src, lengths, bank, bankT);
    k_fc<<<fcBlocks, blk128, 0, stream>>>(bank, fcw_bf, fc_b, S.qs);
    for (int step = 0; step < STEPS; ++step) {
      bf16* ho = (step % 2 == 0) ? hA : hB;
      bf16* hi = (step % 2 == 0) ? hB : hA;  // unused when step==0
      k_gates<<<gateBlocks, blk128, 0, stream>>>(S.qs, hi, ho, cbuf, S.wih, S.whh,
                                                 S.bih, S.bhh, step == 0 ? 1 : 0);
      k_copyh<<<elemBlocks, blk256, 0, stream>>>(ho, S.qs);
      k_e<<<eBlocks, blk128, 0, stream>>>(ho, bank, lengths, ebuf);
      k_softmax<<<smBlocks, blk256, 0, stream>>>(ebuf, abuf);
      k_r<<<rBlocks, blk128, 0, stream>>>(abuf, bankT, S.qs);
    }
  }

  k_offsets<<<1, 32, 0, stream>>>(lengths, offs);
  k_out<<<outBlocks, blk256, 0, stream>>>(qs_s, qs_p, out_w, out_b, lengths, offs, out);
}
